// SuperGlue_57483842290359
// MI455X (gfx1250) — matrix operand order repaired and checked by reading, not compiled, not run on hardware
//
#include <hip/hip_runtime.h>
#include <math.h>

// ---------------------------------------------------------------------------
// Types / constants
// ---------------------------------------------------------------------------
typedef __attribute__((ext_vector_type(16))) _Float16 v16h;
typedef __attribute__((ext_vector_type(8)))  float    v8f;

#define NPTS   2048
#define DDIM   256
#define NHEAD  4
#define DHEAD  64
#define NLAYER 18
#define BN_EPS 1e-5f
#define N1     2049   // NPTS + 1 (transport matrix side)

// ---------------------------------------------------------------------------
// WMMA fragment index maps (CDNA5 ISA 7.12.2, 16-bit A 16x32 / B 32x16)
//   A: lanes 0-15 M=lane, K = e(0..7)->0..7, e(8..15)->16..23 ; lanes 16-31 +8
//   B: lanes 0-15 N=lane, K = same as A ;    lanes 16-31 +8
// ---------------------------------------------------------------------------
__device__ __forceinline__ int kmapA(int e, int lane) {
  return ((e < 8) ? e : (e + 8)) + ((lane & 16) ? 8 : 0);
}
__device__ __forceinline__ int kmapB(int e, int lane) {
  return ((e < 8) ? e : (e + 8)) + ((lane & 16) ? 8 : 0);
}

// ---------------------------------------------------------------------------
// Generic tiled GEMM:  C[m,n] = alpha * sum_k opA(m,k)*opB(k,n) (+ bias[m])
//   opA(m,k) = TA ? A[k*lda+m] : A[m*lda+k]
//   opB(k,n) = TB ? B[n*ldb+k] : B[k*ldb+n]
//
// f32 in/out, f16 WMMA math, f32 accumulation. 64x64 block tile, 4 waves,
// each wave owns a 16x64 row strip (4 accumulators), K-steps of 32.
//
//  * A fragments: thread (wave,lane) consumes exactly one fragment -> built
//    directly in registers from global memory (no LDS).
//  * B fragments: shared by all 4 waves -> staged in LDS pre-swizzled into
//    fragment order (one aligned v16h per slot -> ds_store/load_b128 pairs),
//    double-buffered so each k-step costs a single barrier.
//  * Guard structure: full-tile blocks (block-uniform test) run a guard-free
//    main K loop; K remainder handled by one fully-guarded tail step.
// ---------------------------------------------------------------------------
template<bool TA, bool TB, bool BIAS>
__global__ __launch_bounds__(128) void gemm_wmma_f16(
    const float* __restrict__ A, int lda,
    const float* __restrict__ B, int ldb,
    float* __restrict__ C, int ldc,
    const float* __restrict__ bias,
    int M, int N, int K, float alpha)
{
  // [2 buffers][4 N-subtiles x 32 lanes], 32B fragment slice per slot (8 KB)
  __shared__ v16h BsF[2][128];

  const int tid  = threadIdx.x;
  const int wave = tid >> 5;
  const int lane = tid & 31;
  const int bm   = blockIdx.x * 64;
  const int bn   = blockIdx.y * 64;

  v8f acc[4];
#pragma unroll
  for (int t = 0; t < 4; ++t)
#pragma unroll
    for (int r = 0; r < 8; ++r) acc[t][r] = 0.0f;

  // A fragment row for this thread (fixed across K loop)
  const int  gm  = bm + wave * 16 + (lane & 15);
  const bool mok = (gm < M);
  // B fragment column this thread stages (nt == wave slot)
  const int  gn  = bn + wave * 16 + (lane & 15);
  const bool nok = (gn < N);
  // block-uniform: no M/N guards needed anywhere in this block?
  const bool edge = (bm + 64 > M) || (bn + 64 > N);

  const float* __restrict__ Ap = TA ? (A + gm) : (A + (size_t)gm * lda);
  const float* __restrict__ Bp = TB ? (B + (size_t)gn * ldb) : (B + gn);

  // stage B fragment, one barrier, 4 WMMAs (A fragment reused)
  auto mma_step = [&](const v16h& af, const v16h& bt, int p) {
    BsF[p][tid] = bt;                 // ds_store_b128 x2
    __syncthreads();
#pragma unroll
    for (int nt = 0; nt < 4; ++nt) {
      v16h bf = BsF[p][nt * 32 + lane];   // ds_load_b128 x2
      acc[nt] = __builtin_amdgcn_wmma_f32_16x16x32_f16(
          false, af, false, bf, (short)0, acc[nt], false, false);
    }
  };

  const int kmain = K & ~31;
  int k0 = 0, p = 0;
  if (!edge) {
    // ---------------- guard-free main body ----------------
    for (; k0 < kmain; k0 += 32, p ^= 1) {
      v16h af, bt;
#pragma unroll
      for (int e = 0; e < 16; ++e) {
        int ga = k0 + kmapA(e, lane);
        af[e] = (_Float16)(TA ? Ap[(size_t)ga * lda] : Ap[ga]);
        int gb = k0 + kmapB(e, lane);
        bt[e] = (_Float16)(TB ? Bp[gb] : Bp[(size_t)gb * ldb]);
      }
      mma_step(af, bt, p);
    }
  } else {
    // ---------------- M/N-guarded main body (K unguarded) ----------------
    for (; k0 < kmain; k0 += 32, p ^= 1) {
      v16h af, bt;
#pragma unroll
      for (int e = 0; e < 16; ++e) {
        int ga = k0 + kmapA(e, lane);
        af[e] = (_Float16)(mok ? (TA ? Ap[(size_t)ga * lda] : Ap[ga]) : 0.0f);
        int gb = k0 + kmapB(e, lane);
        bt[e] = (_Float16)(nok ? (TB ? Bp[gb] : Bp[(size_t)gb * ldb]) : 0.0f);
      }
      mma_step(af, bt, p);
    }
  }
  // ---------------- fully-guarded K tail (at most one step) ----------------
  if (k0 < K) {
    v16h af, bt;
#pragma unroll
    for (int e = 0; e < 16; ++e) {
      int ga = k0 + kmapA(e, lane);
      float va = 0.0f;
      if (mok && ga < K) va = TA ? Ap[(size_t)ga * lda] : Ap[ga];
      af[e] = (_Float16)va;
      int gb = k0 + kmapB(e, lane);
      float vb = 0.0f;
      if (nok && gb < K) vb = TB ? Bp[gb] : Bp[(size_t)gb * ldb];
      bt[e] = (_Float16)vb;
    }
    mma_step(af, bt, p);
  }

  // ---- epilogue: C layout VGPR r -> M=r (lanes 0-15) / M=r+8 (16-31) ----
  const int mbase = bm + wave * 16 + ((lane & 16) ? 8 : 0);
#pragma unroll
  for (int nt = 0; nt < 4; ++nt) {
    int n = bn + nt * 16 + (lane & 15);
    if (n < N) {
#pragma unroll
      for (int r = 0; r < 8; ++r) {
        int m = mbase + r;
        if (m < M) {
          float v = alpha * acc[nt][r];
          if (BIAS) v += bias[m];
          C[(size_t)m * ldc + n] = v;
        }
      }
    }
  }
}

// ---------------------------------------------------------------------------
// Block reductions (blockDim.x == 256)
// ---------------------------------------------------------------------------
__device__ __forceinline__ float blk_max(float v) {
  __shared__ float sm[256];
  int tid = threadIdx.x;
  sm[tid] = v; __syncthreads();
  for (int s = 128; s > 0; s >>= 1) {
    if (tid < s) sm[tid] = fmaxf(sm[tid], sm[tid + s]);
    __syncthreads();
  }
  float r = sm[0]; __syncthreads();
  return r;
}
__device__ __forceinline__ float blk_sum(float v) {
  __shared__ float sm[256];
  int tid = threadIdx.x;
  sm[tid] = v; __syncthreads();
  for (int s = 128; s > 0; s >>= 1) {
    if (tid < s) sm[tid] += sm[tid + s];
    __syncthreads();
  }
  float r = sm[0]; __syncthreads();
  return r;
}

// ---------------------------------------------------------------------------
// Elementwise / utility kernels
// ---------------------------------------------------------------------------
__global__ void pack_kenc_k(const float* __restrict__ kpts,
                            const float* __restrict__ scores,
                            float* __restrict__ out, int n) {
  int i = blockIdx.x * blockDim.x + threadIdx.x;
  if (i < n) {
    out[i]         = kpts[2 * i];      // x row (swapaxes)
    out[n + i]     = kpts[2 * i + 1];  // y row
    out[2 * n + i] = scores[i];
  }
}

__global__ void add2_k(const float* __restrict__ a, const float* __restrict__ b,
                       float* __restrict__ c, int n) {
  int i = blockIdx.x * blockDim.x + threadIdx.x;
  if (i < n) c[i] = a[i] + b[i];
}

__global__ void addinp_k(float* __restrict__ a, const float* __restrict__ b, int n) {
  int i = blockIdx.x * blockDim.x + threadIdx.x;
  if (i < n) a[i] += b[i];
}

__global__ void copy_k(const float* __restrict__ a, float* __restrict__ b, int n) {
  int i = blockIdx.x * blockDim.x + threadIdx.x;
  if (i < n) b[i] = a[i];
}

__global__ void zero_k(float* __restrict__ a, int n) {
  int i = blockIdx.x * blockDim.x + threadIdx.x;
  if (i < n) a[i] = 0.0f;
}

// ---------------------------------------------------------------------------
// BatchNorm1d (train mode): per-channel mean/var over N, then scale/shift+ReLU
// ---------------------------------------------------------------------------
__global__ __launch_bounds__(256) void bn_stats_k(const float* __restrict__ x, int n,
                                                  float* __restrict__ mean,
                                                  float* __restrict__ var) {
  int c = blockIdx.x;
  const float* p = x + (size_t)c * n;
  float s = 0.0f, ss = 0.0f;
  for (int j = threadIdx.x; j < n; j += 256) {
    float t = p[j];
    s += t; ss += t * t;
  }
  s  = blk_sum(s);
  ss = blk_sum(ss);
  if (threadIdx.x == 0) {
    float m = s / (float)n;
    mean[c] = m;
    var[c]  = ss / (float)n - m * m;   // biased variance (jnp.var default)
  }
}

__global__ void bn_apply_relu_k(float* __restrict__ x,
                                const float* __restrict__ g,
                                const float* __restrict__ be,
                                const float* __restrict__ mean,
                                const float* __restrict__ var,
                                int npts, int total) {
  int i = blockIdx.x * blockDim.x + threadIdx.x;
  if (i < total) {
    int c = i / npts;
    float t = g[c] * (x[i] - mean[c]) * rsqrtf(var[c] + BN_EPS) + be[c];
    x[i] = fmaxf(t, 0.0f);
  }
}

// ---------------------------------------------------------------------------
// Row softmax (in place). grid = #rows, blockDim = 256.
// ---------------------------------------------------------------------------
__global__ __launch_bounds__(256) void softmax_rows_k(float* __restrict__ S, int cols) {
  float* p = S + (size_t)blockIdx.x * cols;
  float mx = -3.402823466e38f;
  for (int j = threadIdx.x; j < cols; j += 256) mx = fmaxf(mx, p[j]);
  mx = blk_max(mx);
  float s = 0.0f;
  for (int j = threadIdx.x; j < cols; j += 256) {
    float e = __expf(p[j] - mx);
    p[j] = e; s += e;
  }
  s = blk_sum(s);
  float inv = 1.0f / s;
  for (int j = threadIdx.x; j < cols; j += 256) p[j] *= inv;
}

// ---------------------------------------------------------------------------
// Sinkhorn (log-domain optimal transport)
// ---------------------------------------------------------------------------
__global__ void build_Z_k(const float* __restrict__ S, const float* __restrict__ alpha,
                          float* __restrict__ Z) {
  int i = blockIdx.x * blockDim.x + threadIdx.x;
  if (i < N1 * N1) {
    int r = i / N1, c = i % N1;
    Z[i] = (r < NPTS && c < NPTS) ? S[(size_t)r * NPTS + c] : *alpha;
  }
}

// u[i] = log_mu[i] - logsumexp_j(Z[i,j] + v[j])
__global__ __launch_bounds__(256) void sink_u_k(const float* __restrict__ Z,
                                                const float* __restrict__ v,
                                                float* __restrict__ u,
                                                float norm, float logn) {
  int i = blockIdx.x;
  const float* row = Z + (size_t)i * N1;
  float mx = -3.402823466e38f;
  for (int j = threadIdx.x; j < N1; j += 256) mx = fmaxf(mx, row[j] + v[j]);
  mx = blk_max(mx);
  float s = 0.0f;
  for (int j = threadIdx.x; j < N1; j += 256) s += __expf(row[j] + v[j] - mx);
  s = blk_sum(s);
  if (threadIdx.x == 0) {
    float logmu = (i < N1 - 1) ? norm : (logn + norm);
    u[i] = logmu - (mx + __logf(s));
  }
}

// v[j] = log_nu[j] - logsumexp_i(Z[i,j] + u[i])
__global__ __launch_bounds__(256) void sink_v_k(const float* __restrict__ Z,
                                                const float* __restrict__ u,
                                                float* __restrict__ v,
                                                float norm, float logn) {
  int j = blockIdx.x;
  float mx = -3.402823466e38f;
  for (int i = threadIdx.x; i < N1; i += 256) mx = fmaxf(mx, Z[(size_t)i * N1 + j] + u[i]);
  mx = blk_max(mx);
  float s = 0.0f;
  for (int i = threadIdx.x; i < N1; i += 256) s += __expf(Z[(size_t)i * N1 + j] + u[i] - mx);
  s = blk_sum(s);
  if (threadIdx.x == 0) {
    float lognu = (j < N1 - 1) ? norm : (logn + norm);
    v[j] = lognu - (mx + __logf(s));
  }
}

__global__ void transport_out_k(const float* __restrict__ Z, const float* __restrict__ u,
                                const float* __restrict__ v, float* __restrict__ out,
                                float norm) {
  int i = blockIdx.x * blockDim.x + threadIdx.x;
  if (i < N1 * N1) {
    int r = i / N1, c = i % N1;
    out[i] = Z[i] + u[r] + v[c] - norm;
  }
}

// ---------------------------------------------------------------------------
// Host-side GEMM dispatcher
// ---------------------------------------------------------------------------
static void launch_gemm(hipStream_t st, bool ta, bool tb,
                        const float* A, int lda, const float* B, int ldb,
                        float* C, int ldc, const float* bias,
                        int M, int N, int K, float alpha)
{
  dim3 g((unsigned)((M + 63) / 64), (unsigned)((N + 63) / 64));
  dim3 b(128);
#define GG(TA, TB, BB) gemm_wmma_f16<TA, TB, BB><<<g, b, 0, st>>>(A, lda, B, ldb, C, ldc, bias, M, N, K, alpha)
  if (bias) {
    if (!ta && !tb)      GG(false, false, true);
    else if (ta && !tb)  GG(true,  false, true);
    else if (!ta && tb)  GG(false, true,  true);
    else                 GG(true,  true,  true);
  } else {
    if (!ta && !tb)      GG(false, false, false);
    else if (ta && !tb)  GG(true,  false, false);
    else if (!ta && tb)  GG(false, true,  false);
    else                 GG(true,  true,  false);
  }
#undef GG
}

// ---------------------------------------------------------------------------
// kernel_launch
//
// Input flattening assumption: top-level setup_inputs() dict order, nested
// param dicts flattened pytree-style (sorted keys):
//   0 kpts0  1 scores0  2 desc0  3 kpts1  4 scores1  5 desc1
//   kenc (W,b,be,g): 6-9 W0..W3, 10-13 b0..b3, 14-16 be0..be2, 17-19 g0..g2
//   gnn (Wk,Wm,Wq,Wv,bk,bm,bq,bv,mW1,mW2,mb1,mb2,mbe,mg): 20..33
//   34 final_W  35 final_b  36 bin_score
// ---------------------------------------------------------------------------
extern "C" void kernel_launch(void* const* d_in, const int* in_sizes, int n_in,
                              void* d_out, int out_size, void* d_ws, size_t ws_size,
                              hipStream_t stream) {
  (void)in_sizes; (void)n_in; (void)out_size; (void)ws_size;

  const float* kpts0   = (const float*)d_in[0];
  const float* scores0 = (const float*)d_in[1];
  const float* desc0   = (const float*)d_in[2];
  const float* kpts1   = (const float*)d_in[3];
  const float* scores1 = (const float*)d_in[4];
  const float* desc1   = (const float*)d_in[5];
  const float* keW[4]  = {(const float*)d_in[6],  (const float*)d_in[7],
                          (const float*)d_in[8],  (const float*)d_in[9]};
  const float* keb[4]  = {(const float*)d_in[10], (const float*)d_in[11],
                          (const float*)d_in[12], (const float*)d_in[13]};
  const float* kebe[3] = {(const float*)d_in[14], (const float*)d_in[15],
                          (const float*)d_in[16]};
  const float* keg[3]  = {(const float*)d_in[17], (const float*)d_in[18],
                          (const float*)d_in[19]};
  const float* Wk  = (const float*)d_in[20];
  const float* Wm  = (const float*)d_in[21];
  const float* Wq  = (const float*)d_in[22];
  const float* Wv  = (const float*)d_in[23];
  const float* bk  = (const float*)d_in[24];
  const float* bm  = (const float*)d_in[25];
  const float* bq  = (const float*)d_in[26];
  const float* bv  = (const float*)d_in[27];
  const float* mW1 = (const float*)d_in[28];
  const float* mW2 = (const float*)d_in[29];
  const float* mb1 = (const float*)d_in[30];
  const float* mb2 = (const float*)d_in[31];
  const float* mbe = (const float*)d_in[32];
  const float* mg  = (const float*)d_in[33];
  const float* finalW = (const float*)d_in[34];
  const float* finalb = (const float*)d_in[35];
  const float* binsc  = (const float*)d_in[36];

  // ---- workspace layout (f32) ----
  float* Wsp = (float*)d_ws;
  size_t off = 0;
  auto alloc = [&](size_t n) { float* p = Wsp + off; off += n; return p; };
  const size_t DN = (size_t)DDIM * NPTS;   // 524288
  float* f_d0   = alloc(DN);
  float* f_d1   = alloc(DN);
  float* f_q    = alloc(DN);
  float* f_k    = alloc(DN);
  float* f_v    = alloc(DN);
  float* f_msg  = alloc(DN);
  float* f_msgm = alloc(DN);
  float* f_dA   = alloc(DN);
  float* f_dB   = alloc(DN);
  float* f_cat  = alloc(2 * DN);
  float* f_h1   = alloc(2 * DN);
  float* f_mean = alloc(512);
  float* f_var  = alloc(512);
  float* f_pr   = alloc((size_t)NHEAD * NPTS * NPTS);  // 64 MB probs
  float* f_m0   = alloc(DN);
  float* f_m1   = alloc(DN);
  float* f_S    = alloc((size_t)NPTS * NPTS);
  float* f_Z    = alloc((size_t)N1 * N1);
  float* f_u    = alloc(N1);
  float* f_vv   = alloc(N1);
  float* f_ka   = alloc(DN);
  float* f_kb   = alloc(DN);

  auto ew = [](int n) { return dim3((unsigned)((n + 255) / 256)); };

  // ======================= Keypoint encoder ================================
  auto run_kenc = [&](const float* kp, const float* sc, const float* dsc, float* dout) {
    pack_kenc_k<<<ew(NPTS), 256, 0, stream>>>(kp, sc, f_ka, NPTS);
    const int chs[5] = {3, 32, 64, 128, 256};
    const float* cur = f_ka;
    float* nxt = f_kb;
    for (int i = 0; i < 4; ++i) {
      launch_gemm(stream, false, false, keW[i], chs[i], cur, NPTS, nxt, NPTS,
                  keb[i], chs[i + 1], NPTS, chs[i], 1.0f);
      if (i < 3) {
        bn_stats_k<<<chs[i + 1], 256, 0, stream>>>(nxt, NPTS, f_mean, f_var);
        int tot = chs[i + 1] * NPTS;
        bn_apply_relu_k<<<ew(tot), 256, 0, stream>>>(nxt, keg[i], kebe[i],
                                                     f_mean, f_var, NPTS, tot);
      }
      const float* t = cur; cur = nxt; nxt = (float*)t;
    }
    add2_k<<<ew((int)DN), 256, 0, stream>>>(dsc, cur, dout, (int)DN);
  };
  run_kenc(kpts0, scores0, desc0, f_d0);
  run_kenc(kpts1, scores1, desc1, f_d1);

  // ======================= GNN layers ======================================
  auto run_attn = [&](const float* x, const float* src, int li, float* delta) {
    const size_t oW = (size_t)li * DDIM * DDIM, ob = (size_t)li * DDIM;
    const size_t oM1 = (size_t)li * 2 * DDIM * 2 * DDIM;
    const size_t oM2 = (size_t)li * DDIM * 2 * DDIM;
    const size_t o2b = (size_t)li * 2 * DDIM;

    // q/k/v projections (conv1x1)
    launch_gemm(stream, false, false, Wq + oW, DDIM, x,   NPTS, f_q, NPTS,
                bq + ob, DDIM, NPTS, DDIM, 1.0f);
    launch_gemm(stream, false, false, Wk + oW, DDIM, src, NPTS, f_k, NPTS,
                bk + ob, DDIM, NPTS, DDIM, 1.0f);
    launch_gemm(stream, false, false, Wv + oW, DDIM, src, NPTS, f_v, NPTS,
                bv + ob, DDIM, NPTS, DDIM, 1.0f);

    // per-head scores S_h = (Q_h^T K_h) / sqrt(64); channel c = dh*H + h
    for (int h = 0; h < NHEAD; ++h)
      launch_gemm(stream, true, false, f_q + (size_t)h * NPTS, NHEAD * NPTS,
                  f_k + (size_t)h * NPTS, NHEAD * NPTS,
                  f_pr + (size_t)h * NPTS * NPTS, NPTS, nullptr,
                  NPTS, NPTS, DHEAD, 0.125f);
    softmax_rows_k<<<NHEAD * NPTS, 256, 0, stream>>>(f_pr, NPTS);

    // msg_h = V_h * P_h^T  (msg[dh,h,n] layout matches reshape back to D)
    for (int h = 0; h < NHEAD; ++h)
      launch_gemm(stream, false, true, f_v + (size_t)h * NPTS, NHEAD * NPTS,
                  f_pr + (size_t)h * NPTS * NPTS, NPTS,
                  f_msg + (size_t)h * NPTS, NHEAD * NPTS, nullptr,
                  DHEAD, NPTS, NPTS, 1.0f);

    // merge projection
    launch_gemm(stream, false, false, Wm + oW, DDIM, f_msg, NPTS, f_msgm, NPTS,
                bm + ob, DDIM, NPTS, DDIM, 1.0f);

    // MLP([2D,2D,D]) on concat([x, msg])
    copy_k<<<ew((int)DN), 256, 0, stream>>>(x, f_cat, (int)DN);
    copy_k<<<ew((int)DN), 256, 0, stream>>>(f_msgm, f_cat + DN, (int)DN);
    launch_gemm(stream, false, false, mW1 + oM1, 2 * DDIM, f_cat, NPTS, f_h1, NPTS,
                mb1 + o2b, 2 * DDIM, NPTS, 2 * DDIM, 1.0f);
    bn_stats_k<<<2 * DDIM, 256, 0, stream>>>(f_h1, NPTS, f_mean, f_var);
    bn_apply_relu_k<<<ew((int)(2 * DN)), 256, 0, stream>>>(
        f_h1, mg + o2b, mbe + o2b, f_mean, f_var, NPTS, (int)(2 * DN));
    launch_gemm(stream, false, false, mW2 + oM2, 2 * DDIM, f_h1, NPTS, delta, NPTS,
                mb2 + ob, DDIM, NPTS, 2 * DDIM, 1.0f);
  };

  for (int li = 0; li < NLAYER; ++li) {
    const float* s0 = (li & 1) ? f_d1 : f_d0;   // cross on odd layers
    const float* s1 = (li & 1) ? f_d0 : f_d1;
    run_attn(f_d0, s0, li, f_dA);
    run_attn(f_d1, s1, li, f_dB);
    addinp_k<<<ew((int)DN), 256, 0, stream>>>(f_d0, f_dA, (int)DN);
    addinp_k<<<ew((int)DN), 256, 0, stream>>>(f_d1, f_dB, (int)DN);
  }

  // ======================= Final projection + scores =======================
  launch_gemm(stream, false, false, finalW, DDIM, f_d0, NPTS, f_m0, NPTS,
              finalb, DDIM, NPTS, DDIM, 1.0f);
  launch_gemm(stream, false, false, finalW, DDIM, f_d1, NPTS, f_m1, NPTS,
              finalb, DDIM, NPTS, DDIM, 1.0f);
  launch_gemm(stream, true, false, f_m0, NPTS, f_m1, NPTS, f_S, NPTS, nullptr,
              NPTS, NPTS, DDIM, 0.0625f);   // 1/sqrt(256)

  // ======================= Sinkhorn (100 iters, log domain) ================
  build_Z_k<<<ew(N1 * N1), 256, 0, stream>>>(f_S, binsc, f_Z);
  zero_k<<<ew(N1), 256, 0, stream>>>(f_u, N1);
  zero_k<<<ew(N1), 256, 0, stream>>>(f_vv, N1);
  const float norm = -logf(4096.0f);
  const float logn = logf(2048.0f);
  for (int it = 0; it < 100; ++it) {
    sink_u_k<<<N1, 256, 0, stream>>>(f_Z, f_vv, f_u, norm, logn);
    sink_v_k<<<N1, 256, 0, stream>>>(f_Z, f_u, f_vv, norm, logn);
  }
  transport_out_k<<<ew(N1 * N1), 256, 0, stream>>>(f_Z, f_u, f_vv,
                                                   (float*)d_out, norm);
}
